// MultiscaleSelfAttention_71159018160748
// MI455X (gfx1250) — matrix operand order repaired and checked by reading, not compiled, not run on hardware
//
#include <hip/hip_runtime.h>
#include <hip/hip_bf16.h>

#define B_ 4
#define C_ 256
#define N_ 4096
#define D_ 32
#define O_ 320   // 32 (q) + 32 (k) + 256 (v) projection rows

typedef __attribute__((ext_vector_type(16))) _Float16 v16h;
typedef __attribute__((ext_vector_type(8)))  _Float16 v8h;
typedef __attribute__((ext_vector_type(8)))  float    v8f;

// ---- WMMA A-operand loader (16x32 f16 A-layout):
// lane<16 : halves 0..7 = row[k=0..7],  halves 8..15 = row[k=16..23]
// lane>=16: halves 0..7 = row[k=8..15], halves 8..15 = row[k=24..31]
__device__ __forceinline__ v16h load_a16(const _Float16* row, int hoff) {
  v8h lo = *(const v8h*)(row + hoff);
  v8h hi = *(const v8h*)(row + 16 + hoff);
  v16h r;
#pragma unroll
  for (int i = 0; i < 8; ++i) { r[i] = lo[i]; r[i + 8] = hi[i]; }
  return r;
}

__device__ __forceinline__ unsigned pack2h(float a, float b) {
  union { _Float16 h[2]; unsigned u; } p;
  p.h[0] = (_Float16)a; p.h[1] = (_Float16)b;
  return p.u;
}

// ---------------- pooling: p[b][c][wp][hp] = mean over s x s block ----------
__global__ void pool_kernel(const float* __restrict__ x, float* __restrict__ p,
                            int s, int Wp) {
  int idx = blockIdx.x * blockDim.x + threadIdx.x;
  int total = B_ * C_ * Wp * Wp;
  if (idx >= total) return;
  int hp = idx % Wp;
  int wp = (idx / Wp) % Wp;
  int bc = idx / (Wp * Wp);
  const float* src = x + (size_t)bc * 4096 + (size_t)(wp * s) * 64 + hp * s;
  float sum = 0.f;
  for (int i = 0; i < s; ++i)
    for (int j = 0; j < s; ++j) sum += src[i * 64 + j];
  p[idx] = sum * (1.f / (float)(s * s));
}

// select weight row o of the stacked [Wq;Wk;Wv] (each row length 1024)
__device__ __forceinline__ const float* wrow(const float* Wq, const float* Wk,
                                             const float* Wv, int o) {
  if (o < 32)  return Wq + (size_t)o * 1024;
  if (o < 64)  return Wk + (size_t)(o - 32) * 1024;
  return Wv + (size_t)(o - 64) * 1024;
}

// ---- projection GEMM: Y[b][o][col] = sum_c W[o][slice*256+c] * src[b][c][col]
__global__ __launch_bounds__(256) void proj_kernel(
    const float* __restrict__ src, const float* __restrict__ Wq,
    const float* __restrict__ Wk, const float* __restrict__ Wv,
    int slice, int Ns, float* __restrict__ Y) {
  __shared__ float Xl[256][16];
  int b = blockIdx.y;
  int col0 = blockIdx.x * 16;
  int t = threadIdx.x;
  const float* sb = src + (size_t)b * 256 * Ns + col0;
#pragma unroll
  for (int j = 0; j < 16; ++j) {
    int flat = t + 256 * j;
    int r = flat >> 4, c2 = flat & 15;
    Xl[r][c2] = sb[(size_t)r * Ns + c2];
  }
  __syncthreads();
  int col = t & 15;
  int obase = t >> 4;
#pragma unroll 4
  for (int it = 0; it < 20; ++it) {
    int o = obase + it * 16;
    const float* wr = wrow(Wq, Wk, Wv, o) + slice * 256;
    float accv = 0.f;
#pragma unroll 8
    for (int c = 0; c < 256; ++c) accv += wr[c] * Xl[c][col];
    Y[((size_t)b * O_ + o) * Ns + col0 + col] = accv;
  }
}

// ---- combine scales + bias, emit f16 q/k (row-major per token) and v (key-contig)
__global__ __launch_bounds__(256) void combine_kernel(
    const float* __restrict__ Y3, const float* __restrict__ Y2,
    const float* __restrict__ Y4, const float* __restrict__ Y8,
    const float* __restrict__ bq, const float* __restrict__ bk,
    const float* __restrict__ bv,
    _Float16* __restrict__ qh, _Float16* __restrict__ kh,
    _Float16* __restrict__ vh) {
  size_t idx = (size_t)blockIdx.x * blockDim.x + threadIdx.x; // over B*O_*N_
  int i = (int)(idx % N_);
  int o = (int)((idx / N_) % O_);
  int b = (int)(idx / ((size_t)N_ * O_));
  int w = i >> 6, h = i & 63;
  int j2 = ((w >> 1) << 5) + (h >> 1);
  int j4 = ((w >> 2) << 4) + (h >> 2);
  int j8 = ((w >> 3) << 3) + (h >> 3);
  float z = Y3[((size_t)b * O_ + o) * 4096 + i] +
            Y2[((size_t)b * O_ + o) * 1024 + j2] +
            Y4[((size_t)b * O_ + o) * 256  + j4] +
            Y8[((size_t)b * O_ + o) * 64   + j8];
  if (o < 32) {
    z += bq[o];
    qh[((size_t)b * N_ + i) * D_ + o] = (_Float16)z;
  } else if (o < 64) {
    z += bk[o - 32];
    kh[((size_t)b * N_ + i) * D_ + (o - 32)] = (_Float16)z;
  } else {
    z += bv[o - 64];
    vh[((size_t)b * C_ + (o - 64)) * N_ + i] = (_Float16)z;
  }
}

// ------------- two-pass flash attention, one wave per 16-query tile ----------
// Pass 1: exact row max (lane-local partial max, one reduction at the end).
// Pass 2: no running max, no accumulator rescale, no per-block cross-lane ops.
__global__ __launch_bounds__(32) void attn_kernel(
    const _Float16* __restrict__ qh, const _Float16* __restrict__ kh,
    const _Float16* __restrict__ vh, const float* __restrict__ x,
    const float* __restrict__ gamma, float* __restrict__ out) {
  // P staged column-interleaved: Pl[row][2*col + tile], row-major 16x32 f16
  __shared__ _Float16 Pl[16 * 32];
  const int lane = threadIdx.x;
  const int b    = blockIdx.y;
  const int qt   = blockIdx.x;
  const int nlo  = lane & 15;
  const int hi16 = lane >> 4;
  const int hoff = hi16 * 8;

  const _Float16* qrow = qh + ((size_t)b * N_ + qt * 16 + nlo) * D_;
  v16h aq = *(const v16h*)(qrow + hi16 * 16);

  // K B-operand base: col n = lane%16, halves = features hi16*16+0..15 (32B)
  const _Float16* kbase = kh + ((size_t)b * N_ + nlo) * D_ + hi16 * 16;

  // ---------------- pass 1: exact row max ----------------
  float mrow[8];
#pragma unroll
  for (int r = 0; r < 8; ++r) mrow[r] = -3.0e38f;

  {
    v16h bk0 = *(const v16h*)(kbase);
    v16h bk1 = *(const v16h*)(kbase + 16 * D_);
    for (int kb = 0; kb < N_; kb += 32) {
      // prefetch next K block (last prefetch reads into adjacent ws; unused)
      v16h nk0 = *(const v16h*)(kbase + (size_t)(kb + 32) * D_);
      v16h nk1 = *(const v16h*)(kbase + (size_t)(kb + 48) * D_);
      v8f z = {0.f, 0.f, 0.f, 0.f, 0.f, 0.f, 0.f, 0.f};
      v8f s0 = __builtin_amdgcn_wmma_f32_16x16x32_f16(false, aq, false, bk0,
                                                      (short)0, z, false, false);
      v8f s1 = __builtin_amdgcn_wmma_f32_16x16x32_f16(false, aq, false, bk1,
                                                      (short)0, z, false, false);
#pragma unroll
      for (int r = 0; r < 8; ++r)
        mrow[r] = fmaxf(mrow[r], fmaxf(s0[r], s1[r]));
      bk0 = nk0;
      bk1 = nk1;
    }
  }
  // one-time cross-lane reduction within each 16-lane half
#pragma unroll
  for (int r = 0; r < 8; ++r) {
    mrow[r] = fmaxf(mrow[r], __shfl_xor(mrow[r], 1, 32));
    mrow[r] = fmaxf(mrow[r], __shfl_xor(mrow[r], 2, 32));
    mrow[r] = fmaxf(mrow[r], __shfl_xor(mrow[r], 4, 32));
    mrow[r] = fmaxf(mrow[r], __shfl_xor(mrow[r], 8, 32));
  }

  // ---------------- pass 2: accumulate exp(S-M) @ V ----------------
  v8f acc[16];
#pragma unroll
  for (int t = 0; t < 16; ++t)
#pragma unroll
    for (int r = 0; r < 8; ++r) acc[t][r] = 0.f;

  float lsum[8];
#pragma unroll
  for (int r = 0; r < 8; ++r) lsum[r] = 0.f;

  const _Float16* vbase = vh + ((size_t)b * C_ + nlo) * N_ + hi16 * 16;

  v16h bk0 = *(const v16h*)(kbase);
  v16h bk1 = *(const v16h*)(kbase + 16 * D_);
  for (int kb = 0; kb < N_; kb += 32) {
    v16h nk0 = *(const v16h*)(kbase + (size_t)(kb + 32) * D_);
    v16h nk1 = *(const v16h*)(kbase + (size_t)(kb + 48) * D_);
    v8f z = {0.f, 0.f, 0.f, 0.f, 0.f, 0.f, 0.f, 0.f};
    v8f s0 = __builtin_amdgcn_wmma_f32_16x16x32_f16(false, aq, false, bk0,
                                                    (short)0, z, false, false);
    v8f s1 = __builtin_amdgcn_wmma_f32_16x16x32_f16(false, aq, false, bk1,
                                                    (short)0, z, false, false);
    bk0 = nk0;
    bk1 = nk1;

    // exp + lane-local row-sum + packed P store (one b32 per row per lane)
#pragma unroll
    for (int r = 0; r < 8; ++r) {
      float p0 = __expf(s0[r] - mrow[r]);
      float p1 = __expf(s1[r] - mrow[r]);
      lsum[r] += p0 + p1;
      int row = r + hi16 * 8;
      *(unsigned*)(&Pl[row * 32 + 2 * nlo]) = pack2h(p0, p1);
    }
    __syncthreads();

    // reload P as A-operand in contiguous k order: k = 16*hi16 + i
    // (c' = 2k+0 for tile0 cols k, 2k+1 for tile1 cols 16+k)
    v16h pa;
#pragma unroll
    for (int i = 0; i < 16; ++i) pa[i] = Pl[nlo * 32 + 2 * i + hi16];

    // PV: 16 channel chunks; V tiles are contiguous 32B; batch loads by 2
#pragma unroll
    for (int tb = 0; tb < 8; ++tb) {
      v16h bv0 = *(const v16h*)(vbase + (size_t)(2 * tb + 0) * 16 * N_ + kb);
      v16h bv1 = *(const v16h*)(vbase + (size_t)(2 * tb + 1) * 16 * N_ + kb);
      acc[2 * tb + 0] = __builtin_amdgcn_wmma_f32_16x16x32_f16(
          false, pa, false, bv0, (short)0, acc[2 * tb + 0], false, false);
      acc[2 * tb + 1] = __builtin_amdgcn_wmma_f32_16x16x32_f16(
          false, pa, false, bv1, (short)0, acc[2 * tb + 1], false, false);
    }
    __syncthreads();
  }

  // one-time row-sum reduction within each 16-lane half
#pragma unroll
  for (int r = 0; r < 8; ++r) {
    lsum[r] += __shfl_xor(lsum[r], 1, 32);
    lsum[r] += __shfl_xor(lsum[r], 2, 32);
    lsum[r] += __shfl_xor(lsum[r], 4, 32);
    lsum[r] += __shfl_xor(lsum[r], 8, 32);
  }

  // ---- epilogue: out = gamma * acc / l + x ----
  float g = gamma[0];
#pragma unroll
  for (int r = 0; r < 8; ++r) {
    float invl = 1.f / lsum[r];
    int q = qt * 16 + r + hi16 * 8;
#pragma unroll
    for (int t = 0; t < 16; ++t) {
      int ch = t * 16 + nlo;
      size_t idx = ((size_t)b * C_ + ch) * N_ + q;
      out[idx] = g * acc[t][r] * invl + x[idx];
    }
  }
}

extern "C" void kernel_launch(void* const* d_in, const int* in_sizes, int n_in,
                              void* d_out, int out_size, void* d_ws,
                              size_t ws_size, hipStream_t stream) {
  (void)in_sizes; (void)n_in; (void)out_size; (void)ws_size;
  const float* x  = (const float*)d_in[0];
  const float* Wq = (const float*)d_in[1];
  const float* bq = (const float*)d_in[2];
  const float* Wk = (const float*)d_in[3];
  const float* bk = (const float*)d_in[4];
  const float* Wv = (const float*)d_in[5];
  const float* bv = (const float*)d_in[6];
  const float* gm = (const float*)d_in[7];
  float* out = (float*)d_out;

  char* ws = (char*)d_ws;
  size_t off = 0;
  auto alloc = [&](size_t bytes) -> void* {
    void* p = ws + off;
    off += (bytes + 255) & ~(size_t)255;
    return p;
  };
  float* p2 = (float*)alloc((size_t)B_ * C_ * 1024 * 4);
  float* p4 = (float*)alloc((size_t)B_ * C_ * 256 * 4);
  float* p8 = (float*)alloc((size_t)B_ * C_ * 64 * 4);
  float* Y3 = (float*)alloc((size_t)B_ * O_ * 4096 * 4);
  float* Y2 = (float*)alloc((size_t)B_ * O_ * 1024 * 4);
  float* Y4 = (float*)alloc((size_t)B_ * O_ * 256 * 4);
  float* Y8 = (float*)alloc((size_t)B_ * O_ * 64 * 4);
  _Float16* qh = (_Float16*)alloc((size_t)B_ * N_ * D_ * 2);
  _Float16* kh = (_Float16*)alloc((size_t)B_ * N_ * D_ * 2);
  _Float16* vh = (_Float16*)alloc((size_t)B_ * C_ * N_ * 2);

  pool_kernel<<<(B_ * C_ * 1024 + 255) / 256, 256, 0, stream>>>(x, p2, 2, 32);
  pool_kernel<<<(B_ * C_ * 256 + 255) / 256, 256, 0, stream>>>(x, p4, 4, 16);
  pool_kernel<<<(B_ * C_ * 64 + 255) / 256, 256, 0, stream>>>(x, p8, 8, 8);

  proj_kernel<<<dim3(4096 / 16, B_), 256, 0, stream>>>(x,  Wq, Wk, Wv, 3, 4096, Y3);
  proj_kernel<<<dim3(1024 / 16, B_), 256, 0, stream>>>(p2, Wq, Wk, Wv, 0, 1024, Y2);
  proj_kernel<<<dim3(256 / 16,  B_), 256, 0, stream>>>(p4, Wq, Wk, Wv, 1, 256,  Y4);
  proj_kernel<<<dim3(64 / 16,   B_), 256, 0, stream>>>(p8, Wq, Wk, Wv, 2, 64,   Y8);

  combine_kernel<<<(int)(((size_t)B_ * O_ * N_) / 256), 256, 0, stream>>>(
      Y3, Y2, Y4, Y8, bq, bk, bv, qh, kh, vh);

  attn_kernel<<<dim3(N_ / 16, B_), 32, 0, stream>>>(qh, kh, vh, x, gm, out);
}
